// LevelFusion_58145267254108
// MI455X (gfx1250) — matrix operand order repaired and checked by reading, not compiled, not run on hardware
//
#include <hip/hip_runtime.h>

// ---------------- problem constants ----------------
#define TT     2048     // curr time steps
#define DD     512      // feature dim
#define FIN    528      // 8 + 8 + 512 fusion input
#define K1P    544      // FIN padded to multiple of 32
#define MT     128      // rows per workgroup (M tile)
#define NTHR   512      // 16 wave32
#define XS_ST  552      // LDS stride (halves) for X tile  (544 + 8 pad)
#define H1_ST  520      // LDS stride for H1 (512 + 8 pad)
#define H2_ST  264      // LDS stride for H2 (256 + 8 pad)

// ---- LDS overlay layout (bytes). H2 overlays X (X dead after GEMM1). ----
#define SM_XS    0                    // X:  128*552*2 = 141312  (also H2 region)
#define SM_H1    141312               // H1: 128*520*2 = 133120
#define SM_UPS   274432               // 128*8*4  = 4096
#define SM_CURS  278528               // 128*8*4  = 4096
#define SM_FO    282624               // 128*16*4 = 8192
#define SM_GW    290816               // 128*4    = 512
#define SM_GB    291328               // 8*4      = 32
#define SM_SIZE  291360               // ~284.5 KB < 320 KB/WGP

typedef __attribute__((ext_vector_type(2)))  unsigned int v2u;
typedef __attribute__((ext_vector_type(4)))  unsigned int v4u;
typedef __attribute__((ext_vector_type(4)))  float        v4f;
typedef __attribute__((ext_vector_type(8)))  float        v8f;
typedef __attribute__((ext_vector_type(16))) __bf16       v16bf;

union Frag { v4u u[2]; v16bf v; };   // 32B: one WMMA A/B fragment per lane

__device__ __forceinline__ unsigned short f2bf(float x) {
  unsigned int u = __float_as_uint(x);
  u += 0x7FFFu + ((u >> 16) & 1u);          // round-to-nearest-even
  return (unsigned short)(u >> 16);
}
__device__ __forceinline__ unsigned int pack2(float a, float b) {
  return ((unsigned int)f2bf(b) << 16) | (unsigned int)f2bf(a);
}
__device__ __forceinline__ float gelu_f(float x) {
  // x * sigmoid(1.702 x): exp maps to v_exp_f32 trans op
  return x / (1.0f + __expf(-1.702f * x));
}
__device__ __forceinline__ v8f wmma_bf16(const Frag& a, const Frag& b, v8f c) {
  return __builtin_amdgcn_wmma_f32_16x16x32_bf16(false, a.v, false, b.v,
                                                 (short)0, c, false, false);
}

// ---------------- prep: f32 weights -> bf16, transposed [N][K], zero-padded ----
// ws layout (halves): W1T [512][544] | W2T [256][512] | W3T [16][256]
__global__ void prep_weights(const float* __restrict__ W1,
                             const float* __restrict__ W2,
                             const float* __restrict__ W3,
                             unsigned short* __restrict__ ws) {
  const int N1 = 512 * 544, N2 = 256 * 512, N3 = 16 * 256;
  int idx = blockIdx.x * 256 + threadIdx.x;
  if (idx < N1) {
    int n = idx / 544, k = idx % 544;
    ws[idx] = (k < FIN) ? f2bf(W1[k * 512 + n]) : (unsigned short)0;
  } else if (idx < N1 + N2) {
    int j = idx - N1; int n = j / 512, k = j % 512;
    ws[idx] = f2bf(W2[k * 256 + n]);
  } else if (idx < N1 + N2 + N3) {
    int j = idx - N1 - N2; int n = j / 256, k = j % 256;
    ws[idx] = (n < 8) ? f2bf(W3[k * 8 + n]) : (unsigned short)0;
  }
}

// ---------------- fused kernel: 1 workgroup = 128 rows, 16 waves ----------------
__global__ __launch_bounds__(NTHR) void fused_kernel(
    const float* __restrict__ prev, const float* __restrict__ curr,
    const float* __restrict__ feat,
    const float* __restrict__ b1, const float* __restrict__ b2,
    const float* __restrict__ b3,
    const float* __restrict__ Wg, const float* __restrict__ bg,
    const unsigned short* __restrict__ W1T,
    const unsigned short* __restrict__ W2T,
    const unsigned short* __restrict__ W3T,
    float* __restrict__ out) {
  __shared__ __align__(16) unsigned char smem[SM_SIZE];
  unsigned short* Xs  = (unsigned short*)(smem + SM_XS);
  unsigned short* H2s = (unsigned short*)(smem + SM_XS);   // overlays X
  unsigned short* H1s = (unsigned short*)(smem + SM_H1);
  float* ups  = (float*)(smem + SM_UPS);
  float* curs = (float*)(smem + SM_CURS);
  float* fo   = (float*)(smem + SM_FO);
  float* gWs  = (float*)(smem + SM_GW);
  float* gbs  = (float*)(smem + SM_GB);

  const int tid = threadIdx.x;
  const int R0  = blockIdx.x * MT;

  // ---- Stage A: build bf16 X tile in LDS ----
  {                                    // features: 4 threads per row
    int r = tid >> 2, q = tid & 3;
    const v4f* fsrc = (const v4f*)(feat + (size_t)(R0 + r) * DD + q * 128);
    unsigned short* xdst = &Xs[r * XS_ST + 16 + q * 128];
    #pragma unroll 4
    for (int i = 0; i < 32; ++i) {
      v4f f = fsrc[i];
      v2u pk; pk.x = pack2(f.x, f.y); pk.y = pack2(f.z, f.w);
      *(v2u*)(xdst + i * 4) = pk;
    }
  }
  if (tid < MT) {                      // upsample + curr (one row per thread)
    int r = tid, R = R0 + r;
    int b = R >> 11, t = R & (TT - 1);
    float pos = (float)t * (511.0f / 2047.0f);
    int i0 = (int)pos; if (i0 > 510) i0 = 510;
    float w = pos - (float)i0;
    const float* p = prev + ((size_t)b * 512 + i0) * 8;
    v4f a0 = *(const v4f*)(p),     a1 = *(const v4f*)(p + 4);
    v4f d0 = *(const v4f*)(p + 8), d1 = *(const v4f*)(p + 12);
    v4f u0 = a0 + w * (d0 - a0);
    v4f u1 = a1 + w * (d1 - a1);
    const float* cp = curr + (size_t)R * 8;
    v4f c0 = *(const v4f*)(cp), c1 = *(const v4f*)(cp + 4);
    *(v4f*)&ups [r * 8]     = u0;  *(v4f*)&ups [r * 8 + 4] = u1;
    *(v4f*)&curs[r * 8]     = c0;  *(v4f*)&curs[r * 8 + 4] = c1;
    v4u px;
    px.x = pack2(u0.x, u0.y); px.y = pack2(u0.z, u0.w);
    px.z = pack2(u1.x, u1.y); px.w = pack2(u1.z, u1.w);
    *(v4u*)&Xs[r * XS_ST + 0] = px;
    px.x = pack2(c0.x, c0.y); px.y = pack2(c0.z, c0.w);
    px.z = pack2(c1.x, c1.y); px.w = pack2(c1.z, c1.w);
    *(v4u*)&Xs[r * XS_ST + 8] = px;
    v4u z = {0, 0, 0, 0};              // zero K-pad 528..551
    *(v4u*)&Xs[r * XS_ST + 528] = z;
    *(v4u*)&Xs[r * XS_ST + 536] = z;
    *(v4u*)&Xs[r * XS_ST + 544] = z;
  }
  if (tid < 128) gWs[tid] = Wg[tid];
  if (tid < 8)   gbs[tid] = bg[tid];
  __syncthreads();

  const int wv = tid >> 5, l = tid & 31, ln = l & 15, hi = l >> 4;
  const int mtile = wv & 7, nhalf = wv >> 3;
  const v8f vz = {0, 0, 0, 0, 0, 0, 0, 0};

  // ---- GEMM1: H1[128x512] = gelu(X[128x544] @ W1T^T + b1) ----
  {
    const unsigned short* Ar = &Xs[(mtile * 16 + ln) * XS_ST];
    const unsigned short* Bb = W1T + (nhalf * 256 + ln) * K1P + hi * 16;
    v8f acc[16];
    #pragma unroll
    for (int i = 0; i < 16; ++i) acc[i] = vz;
    #pragma unroll 1
    for (int kk = 0; kk < 17; ++kk) {
      int k0 = kk * 32;
      Frag a;
      a.u[0] = *(const v4u*)(Ar + k0 + hi * 16);
      a.u[1] = *(const v4u*)(Ar + k0 + hi * 16 + 8);
      #pragma unroll
      for (int nt = 0; nt < 16; ++nt) {
        const unsigned short* bp = Bb + nt * (16 * K1P) + k0;
        Frag b; b.u[0] = *(const v4u*)(bp); b.u[1] = *(const v4u*)(bp + 8);
        acc[nt] = wmma_bf16(a, b, acc[nt]);
      }
    }
    #pragma unroll
    for (int nt = 0; nt < 16; ++nt) {
      int ncol = nhalf * 256 + nt * 16 + ln;
      float bias = b1[ncol];
      #pragma unroll
      for (int r8 = 0; r8 < 8; ++r8) {
        float x = gelu_f(acc[nt][r8] + bias);
        H1s[(mtile * 16 + hi * 8 + r8) * H1_ST + ncol] = f2bf(x);
      }
    }
  }
  __syncthreads();   // X fully consumed; H2 may now overwrite its region

  // ---- GEMM2: H2[128x256] = gelu(H1 @ W2T^T + b2) ----
  {
    const unsigned short* Ar = &H1s[(mtile * 16 + ln) * H1_ST];
    const unsigned short* Bb = W2T + (nhalf * 128 + ln) * 512 + hi * 16;
    v8f acc[8];
    #pragma unroll
    for (int i = 0; i < 8; ++i) acc[i] = vz;
    #pragma unroll 1
    for (int kk = 0; kk < 16; ++kk) {
      int k0 = kk * 32;
      Frag a;
      a.u[0] = *(const v4u*)(Ar + k0 + hi * 16);
      a.u[1] = *(const v4u*)(Ar + k0 + hi * 16 + 8);
      #pragma unroll
      for (int nt = 0; nt < 8; ++nt) {
        const unsigned short* bp = Bb + nt * (16 * 512) + k0;
        Frag b; b.u[0] = *(const v4u*)(bp); b.u[1] = *(const v4u*)(bp + 8);
        acc[nt] = wmma_bf16(a, b, acc[nt]);
      }
    }
    #pragma unroll
    for (int nt = 0; nt < 8; ++nt) {
      int ncol = nhalf * 128 + nt * 16 + ln;
      float bias = b2[ncol];
      #pragma unroll
      for (int r8 = 0; r8 < 8; ++r8) {
        float x = gelu_f(acc[nt][r8] + bias);
        H2s[(mtile * 16 + hi * 8 + r8) * H2_ST + ncol] = f2bf(x);
      }
    }
  }
  __syncthreads();

  // ---- GEMM3: fo[128x16] = H2 @ W3T^T + b3 (only cols 0..7 valid) ----
  if (wv < 8) {
    const unsigned short* Ar = &H2s[(wv * 16 + ln) * H2_ST];
    const unsigned short* Bb = W3T + ln * 256 + hi * 16;
    v8f c3 = vz;
    #pragma unroll 1
    for (int kk = 0; kk < 8; ++kk) {
      int k0 = kk * 32;
      Frag a;
      a.u[0] = *(const v4u*)(Ar + k0 + hi * 16);
      a.u[1] = *(const v4u*)(Ar + k0 + hi * 16 + 8);
      const unsigned short* bp = Bb + k0;
      Frag b; b.u[0] = *(const v4u*)(bp); b.u[1] = *(const v4u*)(bp + 8);
      c3 = wmma_bf16(a, b, c3);
    }
    float bias = (ln < 8) ? b3[ln] : 0.0f;
    #pragma unroll
    for (int r8 = 0; r8 < 8; ++r8)
      fo[(wv * 16 + hi * 8 + r8) * 16 + ln] = c3[r8] + bias;
  }
  __syncthreads();

  // ---- epilogue: gate + mix, 2 outputs per thread ----
  for (int e = tid; e < MT * 8; e += NTHR) {
    int r = e >> 3, d = e & 7;
    float s = gbs[d];
    #pragma unroll
    for (int j = 0; j < 8; ++j) s += ups [r * 8 + j] * gWs[j * 8 + d];
    #pragma unroll
    for (int j = 0; j < 8; ++j) s += curs[r * 8 + j] * gWs[(8 + j) * 8 + d];
    float g = 1.0f / (1.0f + __expf(-s));
    out[(size_t)(R0 + r) * 8 + d] =
        g * ups[r * 8 + d] + (1.0f - g) * fo[r * 16 + d];
  }
}

// ---------------- host launcher ----------------
extern "C" void kernel_launch(void* const* d_in, const int* in_sizes, int n_in,
                              void* d_out, int out_size, void* d_ws, size_t ws_size,
                              hipStream_t stream) {
  (void)in_sizes; (void)n_in; (void)out_size; (void)ws_size;
  const float* prev = (const float*)d_in[0];
  const float* curr = (const float*)d_in[1];
  const float* feat = (const float*)d_in[2];
  const float* W1   = (const float*)d_in[3];
  const float* b1   = (const float*)d_in[4];
  const float* W2   = (const float*)d_in[5];
  const float* b2   = (const float*)d_in[6];
  const float* W3   = (const float*)d_in[7];
  const float* b3   = (const float*)d_in[8];
  const float* Wg   = (const float*)d_in[9];
  const float* bg   = (const float*)d_in[10];

  unsigned short* ws  = (unsigned short*)d_ws;
  unsigned short* W1T = ws;                    // 512*544 halves
  unsigned short* W2T = ws + 512 * 544;        // 256*512 halves
  unsigned short* W3T = W2T + 256 * 512;       // 16*256 halves

  // (512*544 + 256*512 + 16*256) / 256 = 1616 blocks
  prep_weights<<<1616, 256, 0, stream>>>(W1, W2, W3, ws);

  // 131072 rows / 128 per workgroup = 1024 workgroups
  fused_kernel<<<1024, NTHR, 0, stream>>>(prev, curr, feat, b1, b2, b3,
                                          Wg, bg, W1T, W2T, W3T, (float*)d_out);
}
